// RSSMCore_62732292325747
// MI455X (gfx1250) — matrix operand order repaired and checked by reading, not compiled, not run on hardware
//
#include <hip/hip_runtime.h>

#define DEV __device__ __forceinline__

typedef __bf16 bf16_t;
typedef bf16_t v16bf __attribute__((ext_vector_type(16)));
typedef float  v8f   __attribute__((ext_vector_type(8)));
typedef unsigned int uint4v __attribute__((ext_vector_type(4)));

union Frag32B {
    uint4v       q[2];
    unsigned int u[8];
    v16bf        v;
};

// ---------------- problem dims ----------------
#define NSTEP 64
#define BATCH 32
#define DET   2048
#define STO   512
#define HID   1024
#define EMB   1536
#define G3    6144   // 3*DET

#define NWG   64
#define NTHR  256
#define LDS_BYTES 131072   // 32K dwords: enough for 32xK(K<=2048) bf16 A panel

// ---------------- workspace layout (dword offsets) ----------------
static constexpr size_t O_CNT  = 0;
static constexpr size_t O_WZ   = 16;
static constexpr size_t O_WA   = O_WZ   + (size_t)HID*STO/2;
static constexpr size_t O_WIH  = O_WA   + (size_t)HID*32/2;
static constexpr size_t O_WHH  = O_WIH  + (size_t)G3*HID/2;
static constexpr size_t O_WPH  = O_WHH  + (size_t)G3*DET/2;
static constexpr size_t O_WPE  = O_WPH  + (size_t)HID*DET/2;
static constexpr size_t O_WPOST= O_WPE  + (size_t)HID*EMB/2;
static constexpr size_t O_WPRH = O_WPOST+ (size_t)HID*HID/2;
static constexpr size_t O_WPR  = O_WPRH + (size_t)HID*DET/2;
static constexpr size_t O_ZF   = O_WPR  + (size_t)HID*HID/2;
static constexpr size_t O_AF   = O_ZF   + (size_t)BATCH*STO/2;
static constexpr size_t O_HF   = O_AF   + (size_t)BATCH*32/2;
static constexpr size_t O_ZAF  = O_HF   + (size_t)BATCH*DET/2;
static constexpr size_t O_EF   = O_ZAF  + (size_t)BATCH*HID/2;
static constexpr size_t O_PIF  = O_EF   + (size_t)BATCH*EMB/2;
static constexpr size_t O_XZ   = O_PIF  + (size_t)BATCH*HID/2;
static constexpr size_t O_XA   = O_XZ   + (size_t)BATCH*HID;
static constexpr size_t O_GI   = O_XA   + (size_t)BATCH*HID;
static constexpr size_t O_GH   = O_GI   + (size_t)BATCH*G3;
static constexpr size_t O_XH   = O_GH   + (size_t)BATCH*G3;
static constexpr size_t O_XE   = O_XH   + (size_t)BATCH*HID;
static constexpr size_t O_POST = O_XE   + (size_t)BATCH*HID;
static constexpr size_t O_HST  = O_POST + (size_t)BATCH*HID;
static constexpr size_t O_ZST  = O_HST  + (size_t)BATCH*DET;
static constexpr size_t O_HPF  = O_ZST  + (size_t)BATCH*STO;          // prior H frags (M=2048,K=2048)
static constexpr size_t O_XPR  = O_HPF  + (size_t)2048*2048/2;
static constexpr size_t O_X2F  = O_XPR  + (size_t)2048*1024;
static constexpr size_t O_END  = O_X2F  + (size_t)2048*1024/2;

// ---------------- output layout (float offsets) ----------------
static constexpr size_t OUT_PRI = 0;
static constexpr size_t OUT_POS = OUT_PRI + (size_t)NSTEP*BATCH*HID;
static constexpr size_t OUT_SAM = OUT_POS + (size_t)NSTEP*BATCH*HID;
static constexpr size_t OUT_FEA = OUT_SAM + (size_t)NSTEP*BATCH*STO;
static constexpr size_t OUT_HL  = OUT_FEA + (size_t)NSTEP*BATCH*(DET+STO);
static constexpr size_t OUT_ZL  = OUT_HL  + (size_t)BATCH*DET;

// ---------------- helpers ----------------
DEV unsigned short f2bf(float f) {
    unsigned u = __float_as_uint(f);
    return (unsigned short)((u + 0x7FFFu + ((u >> 16) & 1u)) >> 16);
}

DEV size_t afrag_dw(int Mtiles, int kt, int mt, int lane, int v) {
    return (((size_t)kt * Mtiles + mt) * 32 + lane) * 8 + v;
}

// scatter one (m,k) element into WMMA 16-bit A-fragment layout (ISA 7.12.2)
DEV void afrag_store(unsigned short* base, int Mtiles, int m, int k, unsigned short val) {
    int mt = m >> 4, lr = m & 15, kt = k >> 5, ko = k & 31;
    int grp  = ko >> 3;                     // 0..3
    int lane = lr + ((grp & 1) << 4);       // grp odd -> lanes 16..31
    int v    = ((grp >> 1) << 2) + ((ko & 7) >> 1);
    int slot = ko & 1;
    base[(afrag_dw(Mtiles, kt, mt, lane, v) << 1) + slot] = val;
}

DEV v8f wmma_bf16(const Frag32B& a, const Frag32B& b, v8f c) {
    return __builtin_amdgcn_wmma_f32_16x16x32_bf16(false, a.v, false, b.v, (short)0, c, false, false);
}

DEV void grid_sync(unsigned int* cnt, unsigned int& epoch) {
    __threadfence();
    __syncthreads();
    if (threadIdx.x == 0) {
        epoch += gridDim.x;
        __hip_atomic_fetch_add(cnt, 1u, __ATOMIC_RELEASE, __HIP_MEMORY_SCOPE_AGENT);
        while (__hip_atomic_load(cnt, __ATOMIC_ACQUIRE, __HIP_MEMORY_SCOPE_AGENT) < epoch)
            __builtin_amdgcn_s_sleep(1);
    }
    __syncthreads();
    __threadfence();
}

// GEMM: C[32 x Nt*16] = A(frag,Mtiles=2) * B(packed tiles), bf16 in, f32 acc.
// A panel staged in LDS; wave handles one column tile (both row tiles).
DEV void gemm32(const unsigned int* __restrict__ Af, const unsigned int* __restrict__ Bp,
                float* __restrict__ C, int Nt, int Kt, unsigned int* lds) {
    const int total = Kt * 512;   // dwords in A panel
    __syncthreads();
    for (int i = threadIdx.x * 4; i < total; i += blockDim.x * 4)
        *(uint4v*)(lds + i) = *(const uint4v*)(Af + i);
    __syncthreads();

    const int gw = blockIdx.x * (blockDim.x >> 5) + (threadIdx.x >> 5);
    const int nw = gridDim.x * (blockDim.x >> 5);
    const int l  = threadIdx.x & 31;
    const int ldc = Nt * 16;

    for (int jt = gw; jt < Nt; jt += nw) {
        v8f acc0 = {0,0,0,0,0,0,0,0};
        v8f acc1 = {0,0,0,0,0,0,0,0};
        const unsigned int* bp = Bp + (size_t)jt * Kt * 256 + l * 8;
        for (int kt = 0; kt < Kt; ++kt) {
            Frag32B b, a0, a1;
            b.q[0] = *(const uint4v*)bp;
            b.q[1] = *(const uint4v*)(bp + 4);
            __builtin_prefetch(bp + 256, 0, 0);       // global_prefetch_b8 on next B tile
            const unsigned int* ap = lds + ((kt * 2) * 32 + l) * 8;
            a0.q[0] = *(const uint4v*)ap;       a0.q[1] = *(const uint4v*)(ap + 4);
            a1.q[0] = *(const uint4v*)(ap+256); a1.q[1] = *(const uint4v*)(ap + 260);
            acc0 = wmma_bf16(a0, b, acc0);
            acc1 = wmma_bf16(a1, b, acc1);
            bp += 256;
        }
        int col = jt * 16 + (l & 15);
        int rb  = (l < 16) ? 0 : 8;
        #pragma unroll
        for (int i = 0; i < 8; ++i) {
            C[(size_t)(rb + i)      * ldc + col] = acc0[i];
            C[(size_t)(16 + rb + i) * ldc + col] = acc1[i];
        }
    }
    __syncthreads();
}

// LayerNorm(x1+x2+bias) -> elu -> bf16 A-fragment pack; one WG per batch row
DEV void ln_elu_pack(const float* __restrict__ x1, const float* __restrict__ x2,
                     const float* __restrict__ bias, const float* __restrict__ g,
                     const float* __restrict__ b, unsigned short* __restrict__ frag,
                     float* red) {
    __syncthreads();
    if (blockIdx.x < BATCH) {
        const int row = blockIdx.x, tid = threadIdx.x;
        float vloc[4]; float s = 0.f, ss = 0.f;
        #pragma unroll
        for (int i = 0; i < 4; ++i) {
            int k = tid + i * 256;
            float v = x1[row * HID + k] + x2[row * HID + k] + bias[k];
            vloc[i] = v; s += v; ss += v * v;
        }
        red[tid] = s; red[256 + tid] = ss;
        __syncthreads();
        for (int st = 128; st > 0; st >>= 1) {
            if (tid < st) { red[tid] += red[tid + st]; red[256 + tid] += red[256 + tid + st]; }
            __syncthreads();
        }
        float mean = red[0] * (1.f / HID);
        float var  = red[256] * (1.f / HID) - mean * mean;
        float inv  = rsqrtf(var + 1e-5f);
        #pragma unroll
        for (int i = 0; i < 4; ++i) {
            int k = tid + i * 256;
            float v = (vloc[i] - mean) * inv * g[k] + b[k];
            v = (v > 0.f) ? v : (expf(v) - 1.f);
            afrag_store(frag, 2, row, k, f2bf(v));
        }
    }
    __syncthreads();
}

// ---------------- kernel params ----------------
struct Params {
    const float* embed; const float* action; const unsigned char* reset;
    const float* h0; const float* z0; const float* noises;
    const float* b_z; const float* ln_in_g; const float* ln_in_b;
    const float* bih; const float* bhh;
    const float* b_ph; const float* ln_post_g; const float* ln_post_b;
    const float* b_post;
    float* out;
    unsigned int* ws;
};

__global__ void init_cnt(unsigned int* cnt) { if (threadIdx.x == 0) *cnt = 0u; }

// Pack fp32 weight W[Nout x Kin] (row-major, used as x @ W^T) into bf16 WMMA
// B-fragment tiles (16 cols x 32 k), K zero-padded to Kpad.
__global__ void __launch_bounds__(256) pack_weight(const float* __restrict__ W,
                                                   unsigned int* __restrict__ dst,
                                                   int Nout, int Kin, int Kpad) {
    const int Kt = Kpad >> 5;
    const size_t total = (size_t)Nout * Kpad / 2;
    for (size_t d = (size_t)blockIdx.x * blockDim.x + threadIdx.x; d < total;
         d += (size_t)gridDim.x * blockDim.x) {
        size_t tile = d >> 8;
        int rem  = (int)(d & 255);
        int lane = rem >> 3, v = rem & 7;
        int jt = (int)(tile / Kt), kt = (int)(tile % Kt);
        int n  = jt * 16 + (lane & 15);
        int k0 = kt * 32 + ((v < 4) ? (8 * (lane >> 4) + 2 * v) : (16 + 8 * (lane >> 4) + 2 * (v - 4)));
        float lo = (k0     < Kin) ? W[(size_t)n * Kin + k0]     : 0.f;
        float hi = (k0 + 1 < Kin) ? W[(size_t)n * Kin + k0 + 1] : 0.f;
        dst[d] = (unsigned)f2bf(lo) | ((unsigned)f2bf(hi) << 16);
    }
}

// ---------------- persistent RSSM scan ----------------
__global__ void __launch_bounds__(NTHR, 1) rssm_scan(Params P) {
    extern __shared__ unsigned int lds[];
    unsigned int* ws  = P.ws;
    float*        wsf = (float*)ws;
    unsigned int* cnt = ws + O_CNT;
    float* hst = wsf + O_HST;
    float* zst = wsf + O_ZST;
    unsigned int epoch = 0;
    const int gtid = blockIdx.x * blockDim.x + threadIdx.x;
    const int gsz  = gridDim.x * blockDim.x;

    for (int i = gtid; i < BATCH * DET; i += gsz) hst[i] = P.h0[i];
    for (int i = gtid; i < BATCH * STO; i += gsz) zst[i] = P.z0[i];
    grid_sync(cnt, epoch);

    for (int t = 0; t < NSTEP; ++t) {
        // ---- phase 1: mask + pack z/a/h/e into bf16 A-fragments ----
        {
            unsigned short* zf = (unsigned short*)(ws + O_ZF);
            unsigned short* af = (unsigned short*)(ws + O_AF);
            unsigned short* hf = (unsigned short*)(ws + O_HF);
            unsigned short* ef = (unsigned short*)(ws + O_EF);
            for (int e = gtid; e < BATCH * STO; e += gsz) {
                int m = e >> 9, k = e & 511;
                float mk = P.reset[t * BATCH + m] ? 0.f : 1.f;
                afrag_store(zf, 2, m, k, f2bf(zst[e] * mk));
            }
            for (int e = gtid; e < BATCH * 32; e += gsz) {
                int m = e >> 5, k = e & 31;
                float v = (k < 16) ? P.action[((size_t)t * BATCH + m) * 16 + k] : 0.f;
                afrag_store(af, 2, m, k, f2bf(v));
            }
            for (int e = gtid; e < BATCH * DET; e += gsz) {
                int m = e >> 11, k = e & 2047;
                float mk = P.reset[t * BATCH + m] ? 0.f : 1.f;
                float v = hst[e] * mk;
                hst[e] = v;
                afrag_store(hf, 2, m, k, f2bf(v));
            }
            for (int e = gtid; e < BATCH * EMB; e += gsz) {
                int m = e / EMB, k = e - m * EMB;
                afrag_store(ef, 2, m, k, f2bf(P.embed[(size_t)t * BATCH * EMB + e]));
            }
        }
        grid_sync(cnt, epoch);

        // ---- phase 2: xz = z@Wz^T, xa = a@Wa^T ----
        gemm32(ws + O_ZF, ws + O_WZ, wsf + O_XZ, HID / 16, STO / 32, lds);
        gemm32(ws + O_AF, ws + O_WA, wsf + O_XA, HID / 16, 1, lds);
        grid_sync(cnt, epoch);

        // ---- phase 3: za = elu(LN(xz+xa+b_z)) ----
        ln_elu_pack(wsf + O_XZ, wsf + O_XA, P.b_z, P.ln_in_g, P.ln_in_b,
                    (unsigned short*)(ws + O_ZAF), (float*)lds);
        grid_sync(cnt, epoch);

        // ---- phase 4: gi = za@Wih^T, gh = h@Whh^T ----
        gemm32(ws + O_ZAF, ws + O_WIH, wsf + O_GI, G3 / 16, HID / 32, lds);
        gemm32(ws + O_HF,  ws + O_WHH, wsf + O_GH, G3 / 16, DET / 32, lds);
        grid_sync(cnt, epoch);

        // ---- phase 5: GRU gate -> h_new ----
        {
            const float* gi = wsf + O_GI;
            const float* gh = wsf + O_GH;
            unsigned short* hf = (unsigned short*)(ws + O_HF);
            float* feat = P.out + OUT_FEA + (size_t)t * BATCH * (DET + STO);
            for (int e = gtid; e < BATCH * DET; e += gsz) {
                int m = e >> 11, k = e & 2047;
                size_t bi = (size_t)m * G3;
                float r = 1.f / (1.f + expf(-(gi[bi + k] + P.bih[k] + gh[bi + k] + P.bhh[k])));
                float u = 1.f / (1.f + expf(-(gi[bi + DET + k] + P.bih[DET + k] +
                                              gh[bi + DET + k] + P.bhh[DET + k])));
                float cn = tanhf(gi[bi + 2 * DET + k] + P.bih[2 * DET + k] +
                                 r * (gh[bi + 2 * DET + k] + P.bhh[2 * DET + k]));
                float h  = hst[e];
                float hn = (1.f - u) * cn + u * h;
                hst[e] = hn;
                afrag_store(hf, 2, m, k, f2bf(hn));
                feat[(size_t)m * (DET + STO) + k] = hn;
            }
        }
        grid_sync(cnt, epoch);

        // ---- phase 6: xh = h_new@Wph^T, xe = e@Wpe^T ----
        gemm32(ws + O_HF, ws + O_WPH, wsf + O_XH, HID / 16, DET / 32, lds);
        gemm32(ws + O_EF, ws + O_WPE, wsf + O_XE, HID / 16, EMB / 32, lds);
        grid_sync(cnt, epoch);

        // ---- phase 7: postin = elu(LN(xh+xe+b_ph)) ----
        ln_elu_pack(wsf + O_XH, wsf + O_XE, P.b_ph, P.ln_post_g, P.ln_post_b,
                    (unsigned short*)(ws + O_PIF), (float*)lds);
        grid_sync(cnt, epoch);

        // ---- phase 8: post = postin@Wpost^T ----
        gemm32(ws + O_PIF, ws + O_WPOST, wsf + O_POST, HID / 16, HID / 32, lds);
        grid_sync(cnt, epoch);

        // ---- phase 9: sample, write posts/samples/features ----
        {
            const float* post = wsf + O_POST;
            float* posts_o = P.out + OUT_POS + (size_t)t * BATCH * HID;
            float* sam_o   = P.out + OUT_SAM + (size_t)t * BATCH * STO;
            float* feat    = P.out + OUT_FEA + (size_t)t * BATCH * (DET + STO);
            for (int e = gtid; e < BATCH * HID; e += gsz) {
                int j = e & 1023;
                posts_o[e] = post[e] + P.b_post[j];
            }
            for (int e = gtid; e < BATCH * STO; e += gsz) {
                int m = e >> 9, j = e & 511;
                float mean = post[m * HID + j] + P.b_post[j];
                float sr   = post[m * HID + STO + j] + P.b_post[STO + j];
                float sp   = (sr > 20.f) ? sr : log1pf(expf(sr));
                float sm   = mean + (sp + 0.1f) * P.noises[(size_t)t * BATCH * STO + e];
                sam_o[e] = sm;
                feat[(size_t)m * (DET + STO) + DET + j] = sm;
                zst[e] = sm;
                if (t == NSTEP - 1) P.out[OUT_ZL + e] = sm;
            }
            if (t == NSTEP - 1)
                for (int e = gtid; e < BATCH * DET; e += gsz) P.out[OUT_HL + e] = hst[e];
        }
        grid_sync(cnt, epoch);
    }
}

// ---------------- prior path (batched, after scan) ----------------
__global__ void __launch_bounds__(256) pack_prior_h(const float* __restrict__ feat,
                                                    unsigned int* __restrict__ dst) {
    unsigned short* frag = (unsigned short*)dst;
    const size_t total = (size_t)2048 * 2048;
    for (size_t e = (size_t)blockIdx.x * blockDim.x + threadIdx.x; e < total;
         e += (size_t)gridDim.x * blockDim.x) {
        int m = (int)(e >> 11), k = (int)(e & 2047);
        afrag_store(frag, 128, m, k, f2bf(feat[(size_t)m * (DET + STO) + k]));
    }
}

// generic big-M WMMA GEMM: C[Mt*16 x Nt*16] = A(frag,Mtiles=Mt) * B(tiles) (+bias)
__global__ void __launch_bounds__(256) gemm_big(const unsigned int* __restrict__ Af,
                                                const unsigned int* __restrict__ Bp,
                                                const float* __restrict__ bias,
                                                float* __restrict__ C,
                                                int Mt, int Nt, int Kt, int ldc) {
    const int gw = blockIdx.x * (blockDim.x >> 5) + (threadIdx.x >> 5);
    const int nw = gridDim.x * (blockDim.x >> 5);
    const int l  = threadIdx.x & 31;
    const int Mp = Mt >> 1;
    const int jobs = Mp * Nt;
    for (int job = gw; job < jobs; job += nw) {
        int mp = job / Nt, jt = job - mp * Nt;
        v8f acc0 = {0,0,0,0,0,0,0,0};
        v8f acc1 = {0,0,0,0,0,0,0,0};
        const unsigned int* bp = Bp + (size_t)jt * Kt * 256 + l * 8;
        for (int kt = 0; kt < Kt; ++kt) {
            Frag32B b, a0, a1;
            b.q[0] = *(const uint4v*)bp;
            b.q[1] = *(const uint4v*)(bp + 4);
            __builtin_prefetch(bp + 256, 0, 0);
            const unsigned int* ap = Af + (((size_t)kt * Mt + 2 * mp) * 32 + l) * 8;
            a0.q[0] = *(const uint4v*)ap;         a0.q[1] = *(const uint4v*)(ap + 4);
            a1.q[0] = *(const uint4v*)(ap + 256); a1.q[1] = *(const uint4v*)(ap + 260);
            acc0 = wmma_bf16(a0, b, acc0);
            acc1 = wmma_bf16(a1, b, acc1);
            bp += 256;
        }
        int col = jt * 16 + (l & 15);
        int rb  = (l < 16) ? 0 : 8;
        float bs = bias ? bias[col] : 0.f;
        #pragma unroll
        for (int i = 0; i < 8; ++i) {
            C[(size_t)(mp * 32 + rb + i)      * ldc + col] = acc0[i] + bs;
            C[(size_t)(mp * 32 + 16 + rb + i) * ldc + col] = acc1[i] + bs;
        }
    }
}

// per-row LN(x+bias) -> elu -> bf16 A-fragment pack (one block per row)
__global__ void __launch_bounds__(256) ln_rows(const float* __restrict__ x,
                                               const float* __restrict__ bias,
                                               const float* __restrict__ g,
                                               const float* __restrict__ b,
                                               unsigned short* __restrict__ frag, int Mtiles) {
    __shared__ float red[512];
    const int row = blockIdx.x, tid = threadIdx.x;
    float vloc[4]; float s = 0.f, ss = 0.f;
    #pragma unroll
    for (int i = 0; i < 4; ++i) {
        int k = tid + i * 256;
        float v = x[(size_t)row * HID + k] + bias[k];
        vloc[i] = v; s += v; ss += v * v;
    }
    red[tid] = s; red[256 + tid] = ss;
    __syncthreads();
    for (int st = 128; st > 0; st >>= 1) {
        if (tid < st) { red[tid] += red[tid + st]; red[256 + tid] += red[256 + tid + st]; }
        __syncthreads();
    }
    float mean = red[0] * (1.f / HID);
    float var  = red[256] * (1.f / HID) - mean * mean;
    float inv  = rsqrtf(var + 1e-5f);
    #pragma unroll
    for (int i = 0; i < 4; ++i) {
        int k = tid + i * 256;
        float v = (vloc[i] - mean) * inv * g[k] + b[k];
        v = (v > 0.f) ? v : (expf(v) - 1.f);
        afrag_store(frag, Mtiles, row, k, f2bf(v));
    }
}

// ---------------- host entry ----------------
extern "C" void kernel_launch(void* const* d_in, const int* in_sizes, int n_in,
                              void* d_out, int out_size, void* d_ws, size_t ws_size,
                              hipStream_t stream) {
    unsigned int* ws  = (unsigned int*)d_ws;
    float*        wsf = (float*)d_ws;
    float*        out = (float*)d_out;

    hipFuncSetAttribute(reinterpret_cast<const void*>(rssm_scan),
                        hipFuncAttributeMaxDynamicSharedMemorySize, LDS_BYTES);

    init_cnt<<<1, 32, 0, stream>>>(ws + O_CNT);

    // pack all weights to bf16 WMMA-B tile layout
    pack_weight<<<512, 256, 0, stream>>>((const float*)d_in[6],  ws + O_WZ,   HID, STO,  STO);
    pack_weight<<<512, 256, 0, stream>>>((const float*)d_in[8],  ws + O_WA,   HID, 16,   32);
    pack_weight<<<512, 256, 0, stream>>>((const float*)d_in[11], ws + O_WIH,  G3,  HID,  HID);
    pack_weight<<<512, 256, 0, stream>>>((const float*)d_in[12], ws + O_WHH,  G3,  DET,  DET);
    pack_weight<<<512, 256, 0, stream>>>((const float*)d_in[15], ws + O_WPH,  HID, DET,  DET);
    pack_weight<<<512, 256, 0, stream>>>((const float*)d_in[17], ws + O_WPE,  HID, EMB,  EMB);
    pack_weight<<<512, 256, 0, stream>>>((const float*)d_in[20], ws + O_WPOST,HID, HID,  HID);
    pack_weight<<<512, 256, 0, stream>>>((const float*)d_in[22], ws + O_WPRH, HID, DET,  DET);
    pack_weight<<<512, 256, 0, stream>>>((const float*)d_in[26], ws + O_WPR,  HID, HID,  HID);

    Params P;
    P.embed   = (const float*)d_in[0];
    P.action  = (const float*)d_in[1];
    P.reset   = (const unsigned char*)d_in[2];
    P.h0      = (const float*)d_in[3];
    P.z0      = (const float*)d_in[4];
    P.noises  = (const float*)d_in[5];
    P.b_z     = (const float*)d_in[7];
    P.ln_in_g = (const float*)d_in[9];
    P.ln_in_b = (const float*)d_in[10];
    P.bih     = (const float*)d_in[13];
    P.bhh     = (const float*)d_in[14];
    P.b_ph    = (const float*)d_in[16];
    P.ln_post_g = (const float*)d_in[18];
    P.ln_post_b = (const float*)d_in[19];
    P.b_post  = (const float*)d_in[21];
    P.out     = out;
    P.ws      = ws;

    rssm_scan<<<NWG, NTHR, LDS_BYTES, stream>>>(P);

    // batched prior over all h
    pack_prior_h<<<512, 256, 0, stream>>>(out + OUT_FEA, ws + O_HPF);
    gemm_big<<<512, 256, 0, stream>>>(ws + O_HPF, ws + O_WPRH, nullptr,
                                      wsf + O_XPR, 128, 64, 64, 1024);
    ln_rows<<<2048, 256, 0, stream>>>(wsf + O_XPR, (const float*)d_in[23],
                                      (const float*)d_in[24], (const float*)d_in[25],
                                      (unsigned short*)(ws + O_X2F), 128);
    gemm_big<<<512, 256, 0, stream>>>(ws + O_X2F, ws + O_WPR, (const float*)d_in[27],
                                      out + OUT_PRI, 128, 64, 32, 1024);
}
